// QuantumVQCHead_44504451121770
// MI455X (gfx1250) — matrix operand order repaired and checked by reading, not compiled, not run on hardware
//
#include <hip/hip_runtime.h>
#include <math.h>

typedef __attribute__((ext_vector_type(16))) _Float16 v16h;
typedef __attribute__((ext_vector_type(8)))  float    v8f;
typedef __attribute__((ext_vector_type(4)))  float    f4v;

#define NQ   8
#define NST  256           // 2^8 states
#define HALF_PI_F 1.57079632679f
#define PI_F      3.14159265f

// -------------------------------------------------------------------------
// Workspace layout (d_ws):
//   [0            , B*16*4)          : cs  float[B][8][2]  (cos(a/2), sin(a/2))
//   [B*16*4       , +256KB)          : U fragments, _Float16
//        indexed ((c*8 + kt)*16 + nt)*512 + lane*16 + e   (c=0 Re, c=1 Im)
//        i.e. per (c,kt,nt) a 32-lane x 32-byte WMMA B-fragment
// -------------------------------------------------------------------------

// =========================================================================
// Stage A: build the batch-independent circuit unitary U (everything after
// the initial RY layer).  Block t evolves basis state e_t (column t of U)
// in LDS, then scatters it into pre-swizzled f16 WMMA B-fragments:
//   B[k][n] = U[s = nt*16+n][t = kt*32+k]   ( psi = U * phi  =>  B = U^T)
// B-fragment layout (16-bit, 32x16): lane = ((k>>3)&1)*16 + n, elem = (k&7) + 8*(k>>4).
// =========================================================================
__global__ void __launch_bounds__(256)
build_unitary(const float* __restrict__ wts, _Float16* __restrict__ ufrag) {
    __shared__ float re[NST];
    __shared__ float im[NST];
    const int t = blockIdx.x;     // which basis column
    const int s = threadIdx.x;    // which amplitude this thread owns

    re[s] = (s == t) ? 1.0f : 0.0f;
    im[s] = 0.0f;
    __syncthreads();

    for (int l = 0; l < 2; ++l) {
        for (int w = 0; w < NQ; ++w) {
            const int mask = 1 << (7 - w);   // qubit 0 = MSB of flat index
            float th, c, sn;
            // ---- RX(theta) : [[c, -i s],[-i s, c]] ----
            th = wts[(l * NQ + w) * 3 + 0];
            c = cosf(0.5f * th); sn = sinf(0.5f * th);
            if (!(s & mask)) {
                const int p = s | mask;
                float x0r = re[s], x0i = im[s], x1r = re[p], x1i = im[p];
                re[s] = c * x0r + sn * x1i;  im[s] = c * x0i - sn * x1r;
                re[p] = c * x1r + sn * x0i;  im[p] = c * x1i - sn * x0r;
            }
            __syncthreads();
            // ---- RY(theta) : [[c, -s],[s, c]] ----
            th = wts[(l * NQ + w) * 3 + 1];
            c = cosf(0.5f * th); sn = sinf(0.5f * th);
            if (!(s & mask)) {
                const int p = s | mask;
                float x0r = re[s], x0i = im[s], x1r = re[p], x1i = im[p];
                re[s] = c * x0r - sn * x1r;  im[s] = c * x0i - sn * x1i;
                re[p] = sn * x0r + c * x1r;  im[p] = sn * x0i + c * x1i;
            }
            __syncthreads();
            // ---- RZ(theta) : diag(e^{-i t/2}, e^{+i t/2}) ----
            th = wts[(l * NQ + w) * 3 + 2];
            c = cosf(0.5f * th); sn = sinf(0.5f * th);
            {
                const float ei = (s & mask) ? sn : -sn;   // imag part of phase
                float r = re[s], i = im[s];
                re[s] = r * c - i * ei;
                im[s] = r * ei + i * c;
            }
            __syncthreads();
        }
        // ---- CNOT ring ----
        for (int w = 0; w < NQ; ++w) {
            const int cmask = 1 << (7 - w);
            const int tmask = 1 << (7 - ((w + 1) & 7));
            if ((s & cmask) && !(s & tmask)) {
                const int p = s ^ tmask;
                float tr = re[s], ti = im[s];
                re[s] = re[p];  im[s] = im[p];
                re[p] = tr;     im[p] = ti;
            }
            __syncthreads();
        }
    }

    // Scatter column t into B-fragments (f16), Re (c=0) and Im (c=1).
    const int kt = t >> 5, k = t & 31;
    const int laneHalf = ((k >> 3) & 1) * 16;        // +n gives lane
    const int eoff = (k & 7) + 8 * (k >> 4);         // table order element
    const int c  = s & 1;
    const int nt = (s >> 1) & 15;
    const int n0 = s >> 5;                           // 0..7: handles n0, n0+8
    const float* src = c ? im : re;
    #pragma unroll
    for (int q = 0; q < 2; ++q) {
        const int n = n0 + 8 * q;
        const float v = src[nt * 16 + n];
        const size_t off = (size_t)((c * 8 + kt) * 16 + nt) * 512
                         + (size_t)(laneHalf + n) * 16 + eoff;
        ufrag[off] = (_Float16)v;
    }
}

// =========================================================================
// Stage B: a = tanh(h @ W^T + b) * pi/2, clipped; emit (cos(a/2), sin(a/2)).
// Memory-bound: streams 128 MB of h exactly once -> non-temporal loads so
// the hot U-fragment / cs working set stays resident in L2.
// =========================================================================
__global__ void __launch_bounds__(256)
pre_angles(const float* __restrict__ h, const float* __restrict__ W,
           const float* __restrict__ bias, float* __restrict__ cs) {
    const int g = blockIdx.x * 256 + threadIdx.x;   // g = b*8 + w
    const int b = g >> 3, w = g & 7;
    float acc = bias[w];
    const f4v* h4 = (const f4v*)h + (size_t)b * 128;
    const f4v* w4 = (const f4v*)W + (size_t)w * 128;
    #pragma unroll 4
    for (int k = 0; k < 128; ++k) {
        f4v a = __builtin_nontemporal_load(&h4[k]);   // stream: NT hint
        f4v q = w4[k];                                // tiny, cache normally
        acc += a.x * q.x + a.y * q.y + a.z * q.z + a.w * q.w;
    }
    float t = tanhf(acc) * HALF_PI_F;
    if (t != t) t = 0.0f;                       // nan_to_num
    t = fminf(fmaxf(t, -PI_F), PI_F);           // clip (no-op for finite tanh)
    cs[(size_t)g * 2 + 0] = cosf(0.5f * t);
    cs[(size_t)g * 2 + 1] = sinf(0.5f * t);
}

// =========================================================================
// Stage C: per 32-batch tile (two 16-row m-tiles to amortize B traffic):
//   phi[m][t] = prod_w cs[m][w][bit_w(t)]  (real product state, built in
//   A-fragment LDS layout per ISA 7.12.2), then
//   psi_re = phi x Ure^T, psi_im = phi x Uim^T  via v_wmma_f32_16x16x32_f16,
//   probs = psi_re^2 + psi_im^2, z_w = sum_s (+-) probs, out = z/||z||.
// 4 waves/block, each wave owns 4 N-tiles x 2 m-tiles x Re/Im = 16 accums.
// Each B fragment loaded from L2 feeds 2 WMMAs (one per m-tile).
// =========================================================================
__global__ void __launch_bounds__(128)
vqc_wmma(const float* __restrict__ cs, const _Float16* __restrict__ ufrag,
         float* __restrict__ out) {
    __shared__ __align__(32) _Float16 afrag[2][8][32][16]; // 16 KB: A frags
    __shared__ float csl[32][8][2];                        // 2 KB
    __shared__ float probs[32][NST];                       // 32 KB
    __shared__ float zl[32][8];

    const int tid = threadIdx.x;
    const int b0  = blockIdx.x * 32;

    // cos/sin for this tile's 32 batch rows
    for (int i = tid; i < 512; i += 128) {
        const int m = i >> 4, rest = i & 15;              // rest = w*2 + comp
        csl[m][rest >> 1][rest & 1] = cs[(size_t)(b0 + m) * 16 + rest];
    }
    __syncthreads();

    // Build phi directly in 16-bit A-fragment layout:
    // lane = h*16 + m,  h = (k>>3)&1,  j = ((k&15)&7)>>1 + 4*(k>>4), sub = k&1
    for (int i = tid; i < 8192; i += 128) {
        const int m = i >> 8, kk = i & 255;
        float p = 1.0f;
        #pragma unroll
        for (int w = 0; w < 8; ++w)
            p *= csl[m][w][(kk >> (7 - w)) & 1];
        const int kt = kk >> 5, k = kk & 31;
        const int hi = k >> 4, within = k & 15;
        const int hh = within >> 3;
        const int j  = ((within & 7) >> 1) + 4 * hi;
        afrag[m >> 4][kt][hh * 16 + (m & 15)][2 * j + (k & 1)] = (_Float16)p;
    }
    __syncthreads();

    const int wave = tid >> 5, lane = tid & 31;
    v8f accR[2][4], accI[2][4];
    #pragma unroll
    for (int mh = 0; mh < 2; ++mh)
        #pragma unroll
        for (int i = 0; i < 4; ++i) { accR[mh][i] = {}; accI[mh][i] = {}; }

    for (int kt = 0; kt < 8; ++kt) {
        v16h av[2];
        #pragma unroll
        for (int mh = 0; mh < 2; ++mh)
            av[mh] = *(const v16h*)&afrag[mh][kt][lane][0];
        #pragma unroll
        for (int i = 0; i < 4; ++i) {
            const int nt = wave * 4 + i;
            const v16h bR = *(const v16h*)(ufrag + (size_t)((0 + kt) * 16 + nt) * 512 + lane * 16);
            const v16h bI = *(const v16h*)(ufrag + (size_t)((8 + kt) * 16 + nt) * 512 + lane * 16);
            #pragma unroll
            for (int mh = 0; mh < 2; ++mh) {
                accR[mh][i] = __builtin_amdgcn_wmma_f32_16x16x32_f16(
                    false, av[mh], false, bR, (short)0, accR[mh][i], false, false);
                accI[mh][i] = __builtin_amdgcn_wmma_f32_16x16x32_f16(
                    false, av[mh], false, bI, (short)0, accI[mh][i], false, false);
            }
        }
    }

    // |psi|^2 into LDS.  C-layout: VGPR r, lane L -> row m=r+8*(L>>4), col n=L&15
    {
        const int n = lane & 15, hh = lane >> 4;
        #pragma unroll
        for (int mh = 0; mh < 2; ++mh)
            #pragma unroll
            for (int i = 0; i < 4; ++i) {
                const int nt = wave * 4 + i;
                #pragma unroll
                for (int r = 0; r < 8; ++r) {
                    const float pr = accR[mh][i][r], pi = accI[mh][i][r];
                    probs[mh * 16 + r + 8 * hh][nt * 16 + n] = pr * pr + pi * pi;
                }
            }
    }
    __syncthreads();

    // z_w = sum_s (1 - 2*bit_w(s)) * probs[m][s]; each thread does 2 rows
    {
        const int mbase = tid >> 3, w = tid & 7;
        #pragma unroll
        for (int mh = 0; mh < 2; ++mh) {
            const int m = mbase + 16 * mh;
            float z = 0.0f;
            for (int s = 0; s < NST; ++s) {
                const float sign = ((s >> (7 - w)) & 1) ? -1.0f : 1.0f;
                z += sign * probs[m][s];
            }
            zl[m][w] = z;
        }
    }
    __syncthreads();
    {
        const int mbase = tid >> 3, w = tid & 7;
        #pragma unroll
        for (int mh = 0; mh < 2; ++mh) {
            const int m = mbase + 16 * mh;
            float nrm = 0.0f;
            #pragma unroll
            for (int q = 0; q < 8; ++q) nrm += zl[m][q] * zl[m][q];
            nrm = fmaxf(sqrtf(nrm), 1e-12f);
            out[(size_t)(b0 + m) * 8 + w] = zl[m][w] / nrm;
        }
    }
}

// =========================================================================
extern "C" void kernel_launch(void* const* d_in, const int* in_sizes, int n_in,
                              void* d_out, int out_size, void* d_ws, size_t ws_size,
                              hipStream_t stream) {
    const float* h     = (const float*)d_in[0];   // [B, 512]
    const float* W_pre = (const float*)d_in[1];   // [8, 512]
    const float* b_pre = (const float*)d_in[2];   // [8]
    const float* wts   = (const float*)d_in[3];   // [2, 8, 3]
    float* out = (float*)d_out;                   // [B, 8]

    const int B = in_sizes[0] / 512;              // 65536

    float*     cs    = (float*)d_ws;                                  // B*16 floats
    _Float16*  ufrag = (_Float16*)((char*)d_ws + (size_t)B * 16 * 4); // 256 KB

    // Stage A: build batch-independent unitary -> swizzled f16 B-fragments
    build_unitary<<<NST, NST, 0, stream>>>(wts, ufrag);

    // Stage B: projection + tanh/clip -> (cos(a/2), sin(a/2))
    pre_angles<<<(B * 8) / 256, 256, 0, stream>>>(h, W_pre, b_pre, cs);

    // Stage C: WMMA state evolution + <Z_w> + normalize (32 batches/block)
    vqc_wmma<<<B / 32, 128, 0, stream>>>(cs, ufrag, out);
}
